// VulnerabilityGNN_44384192037112
// MI455X (gfx1250) — matrix operand order repaired and checked by reading, not compiled, not run on hardware
//
#include <hip/hip_runtime.h>
#include <hip/hip_bf16.h>
#include <cmath>

#define N_NODES 50000
#define N_EDGES 800000
#define F_IN    128
#define H_DIM   64
#define N_HEADS 4
#define L_GCN   3
#define G_GRAPH 64
#define N_CLS   2
#define BN_EPS  1e-5f

typedef __attribute__((ext_vector_type(2)))  float    v2f;
typedef __attribute__((ext_vector_type(8)))  float    v8f;
typedef __attribute__((ext_vector_type(16))) _Float16 v16h;

#if __has_builtin(__builtin_amdgcn_wmma_f32_16x16x4_f32)
#define USE_F32_WMMA 1
#endif

// ---------------------------------------------------------------------------
// WMMA GEMM: C[M,N] = act(A[M,K] @ B[K,N] + bias).  One wave = one 16x16 tile.
// M multiple of 16, K multiple of 4 (f32 path) / 32 (f16 fallback), N mult 16.
// ---------------------------------------------------------------------------
__global__ void gemm_wmma_kernel(const float* __restrict__ A,
                                 const float* __restrict__ B,
                                 const float* __restrict__ bias,
                                 float* __restrict__ C,
                                 int M, int K, int Nout, int relu) {
  const int wave = threadIdx.x >> 5;
  const int lane = threadIdx.x & 31;
  const int numTilesM = M >> 4;
  const int tileM = blockIdx.x * (blockDim.x >> 5) + wave;
  if (tileM >= numTilesM) return;            // uniform per wave -> EXEC stays full
  const int tileN = blockIdx.y << 4;
  const int half = lane >> 4;                // 0 or 1
  const int l16  = lane & 15;

  v8f acc = {0.f,0.f,0.f,0.f,0.f,0.f,0.f,0.f};

#ifdef USE_F32_WMMA
  // A(16x4): lanes 0-15 -> M=lane, K = k0+{0,1}; lanes 16-31 -> K = k0+{2,3}
  const float* Arow = A + (size_t)(tileM * 16 + l16) * K + half * 2;
  for (int k0 = 0; k0 < K; k0 += 4) {
    v2f a, b;
    a.x = Arow[k0];
    a.y = Arow[k0 + 1];
    const float* Bp = B + (size_t)(k0 + half * 2) * Nout + tileN + l16;
    b.x = Bp[0];
    b.y = Bp[Nout];
    acc = __builtin_amdgcn_wmma_f32_16x16x4_f32(false, a, false, b,
                                                (short)0, acc, false, false);
  }
#else
  // Fallback: f16 inputs, f32 accumulate (codegen-confirmed builtin).
  const float* Arow = A + (size_t)(tileM * 16 + l16) * K;
  const int col = tileN + l16;
  for (int k0 = 0; k0 < K; k0 += 32) {
    v16h a, b;
    for (int i = 0; i < 8; ++i) {
      a[i]     = (_Float16)Arow[k0 + half * 8 + i];
      a[8 + i] = (_Float16)Arow[k0 + 16 + half * 8 + i];
    }
    for (int i = 0; i < 16; ++i)
      b[i] = (_Float16)B[(size_t)(k0 + (i < 8 ? half * 8 + i : 16 + half * 8 + (i - 8))) * Nout + col];
    acc = __builtin_amdgcn_wmma_f32_16x16x32_f16(false, a, false, b,
                                                 (short)0, acc, false, false);
  }
#endif

  const int colO = tileN + l16;
  const float bv = bias ? bias[colO] : 0.0f;
  for (int r = 0; r < 8; ++r) {
    const int row = tileM * 16 + r + half * 8;   // C/D layout: VGPR r -> M = r (+8)
    float v = acc[r] + bv;
    if (relu) v = fmaxf(v, 0.0f);
    C[(size_t)row * Nout + colO] = v;
  }
}

// ---------------------------------------------------------------------------
// Helpers
// ---------------------------------------------------------------------------
__device__ inline float leaky02(float x) { return x > 0.0f ? x : 0.2f * x; }

__device__ inline void atomicMaxFloat(float* addr, float value) {
  // sign-split trick: correct total order for mixed-sign IEEE floats
  if (value >= 0.0f) atomicMax((int*)addr, __float_as_int(value));
  else               atomicMin((unsigned int*)addr, __float_as_uint(value));
}

// ---------------------------------------------------------------------------
// Degree / normalization  (deg = in-degree + 1 self loop; dinv = rsqrt(deg))
// ---------------------------------------------------------------------------
__global__ void deg_init(float* deg) {
  int i = blockIdx.x * blockDim.x + threadIdx.x;
  if (i < N_NODES) deg[i] = 1.0f;
}
__global__ void deg_edges(const int* __restrict__ dst, float* deg) {
  int e = blockIdx.x * blockDim.x + threadIdx.x;
  if (e < N_EDGES) atomicAdd(&deg[dst[e]], 1.0f);
}
__global__ void deg_finalize(float* deg) {
  int i = blockIdx.x * blockDim.x + threadIdx.x;
  if (i < N_NODES) deg[i] = rsqrtf(deg[i]);
}

// ---------------------------------------------------------------------------
// GCN: y = hw * dinv^2 + b   then  y[dst] += hw[src] * dinv[src]*dinv[dst]
// ---------------------------------------------------------------------------
__global__ void gcn_self_init(const float* __restrict__ hw,
                              const float* __restrict__ dinv,
                              const float* __restrict__ b,
                              float* __restrict__ y) {
  int idx = blockIdx.x * blockDim.x + threadIdx.x;
  if (idx >= N_NODES * H_DIM) return;
  int n = idx >> 6, c = idx & 63;
  float di = dinv[n];
  y[idx] = hw[idx] * di * di + b[c];
}

__global__ void gcn_edge_scatter(const int* __restrict__ src,
                                 const int* __restrict__ dst,
                                 const float* __restrict__ dinv,
                                 const float* __restrict__ hw,
                                 float* __restrict__ y) {
  int idx = blockIdx.x * blockDim.x + threadIdx.x;   // E * (H/4)
  if (idx >= N_EDGES * (H_DIM / 4)) return;
  int e  = idx >> 4;
  int c4 = (idx & 15) << 2;
  int s = src[e], d = dst[e];
  float norm = dinv[s] * dinv[d];
  float4 v = *(const float4*)(hw + (size_t)s * H_DIM + c4);
  float* yp = y + (size_t)d * H_DIM + c4;
  atomicAdd(yp + 0, v.x * norm);
  atomicAdd(yp + 1, v.y * norm);
  atomicAdd(yp + 2, v.z * norm);
  atomicAdd(yp + 3, v.w * norm);
}

// ---------------------------------------------------------------------------
// BatchNorm (training-mode stats over node dim) + activation + residual
// t = y*scale + bias[c]  (scale/bias fold GAT head-mean + bg)
// ---------------------------------------------------------------------------
__global__ void bn_stats_zero(float* stats) {
  int i = threadIdx.x;
  if (i < 2 * H_DIM) stats[i] = 0.0f;
}

__global__ void bn_stats(const float* __restrict__ y, float scale,
                         const float* __restrict__ bias, float* stats) {
  __shared__ float ssum[256], ssq[256];
  int c = threadIdx.x & 63;
  int g = threadIdx.x >> 6;
  float bv = bias ? bias[c] : 0.0f;
  float s = 0.0f, q = 0.0f;
  for (int n = blockIdx.x * 4 + g; n < N_NODES; n += gridDim.x * 4) {
    float t = y[(size_t)n * H_DIM + c] * scale + bv;
    s += t; q += t * t;
  }
  ssum[threadIdx.x] = s; ssq[threadIdx.x] = q;
  __syncthreads();
  if (threadIdx.x < 64) {
    s = ssum[threadIdx.x] + ssum[threadIdx.x + 64] +
        ssum[threadIdx.x + 128] + ssum[threadIdx.x + 192];
    q = ssq[threadIdx.x] + ssq[threadIdx.x + 64] +
        ssq[threadIdx.x + 128] + ssq[threadIdx.x + 192];
    atomicAdd(&stats[c], s);
    atomicAdd(&stats[H_DIM + c], q);
  }
}

__global__ void bn_act_residual(const float* __restrict__ y, float scale,
                                const float* __restrict__ bias,
                                const float* __restrict__ stats,
                                const float* __restrict__ gamma,
                                const float* __restrict__ beta,
                                float* __restrict__ h, int act /*0 relu,1 elu*/) {
  int idx = blockIdx.x * blockDim.x + threadIdx.x;
  if (idx >= N_NODES * H_DIM) return;
  int c = idx & 63;
  float mu  = stats[c] * (1.0f / N_NODES);
  float var = stats[H_DIM + c] * (1.0f / N_NODES) - mu * mu;
  float t = y[idx] * scale + (bias ? bias[c] : 0.0f);
  float v = (t - mu) * rsqrtf(var + BN_EPS) * gamma[c] + beta[c];
  v = (act == 0) ? fmaxf(v, 0.0f) : (v > 0.0f ? v : expf(v) - 1.0f);
  h[idx] = v + h[idx];
}

// ---------------------------------------------------------------------------
// GAT attention
// ---------------------------------------------------------------------------
__global__ void gat_logits(const float* __restrict__ hg,
                           const float* __restrict__ att_s,
                           const float* __restrict__ att_d,
                           float* __restrict__ a_s, float* __restrict__ a_d) {
  int idx = blockIdx.x * blockDim.x + threadIdx.x;   // N * HEADS
  if (idx >= N_NODES * N_HEADS) return;
  int n = idx >> 2, hd = idx & 3;
  const float* hp = hg + (size_t)n * (N_HEADS * H_DIM) + hd * H_DIM;
  const float* as = att_s + hd * H_DIM;
  const float* ad = att_d + hd * H_DIM;
  float ss = 0.0f, sd = 0.0f;
  for (int c = 0; c < H_DIM; ++c) { float v = hp[c]; ss += v * as[c]; sd += v * ad[c]; }
  a_s[idx] = ss; a_d[idx] = sd;
}

__global__ void gat_m_init(const float* __restrict__ a_s, const float* __restrict__ a_d,
                           float* __restrict__ m) {
  int idx = blockIdx.x * blockDim.x + threadIdx.x;
  if (idx < N_NODES * N_HEADS) m[idx] = leaky02(a_s[idx] + a_d[idx]); // self loop
}
__global__ void gat_m_edges(const int* __restrict__ src, const int* __restrict__ dst,
                            const float* __restrict__ a_s, const float* __restrict__ a_d,
                            float* __restrict__ m) {
  int idx = blockIdx.x * blockDim.x + threadIdx.x;   // E * HEADS
  if (idx >= N_EDGES * N_HEADS) return;
  int e = idx >> 2, hd = idx & 3;
  float lg = leaky02(a_s[src[e] * N_HEADS + hd] + a_d[dst[e] * N_HEADS + hd]);
  atomicMaxFloat(&m[dst[e] * N_HEADS + hd], lg);
}
__global__ void gat_s_init(const float* __restrict__ a_s, const float* __restrict__ a_d,
                           const float* __restrict__ m, float* __restrict__ s) {
  int idx = blockIdx.x * blockDim.x + threadIdx.x;
  if (idx >= N_NODES * N_HEADS) return;
  s[idx] = expf(leaky02(a_s[idx] + a_d[idx]) - m[idx]);  // self-loop term
}
__global__ void gat_s_edges(const int* __restrict__ src, const int* __restrict__ dst,
                            const float* __restrict__ a_s, const float* __restrict__ a_d,
                            const float* __restrict__ m, float* __restrict__ s) {
  int idx = blockIdx.x * blockDim.x + threadIdx.x;
  if (idx >= N_EDGES * N_HEADS) return;
  int e = idx >> 2, hd = idx & 3;
  int di = dst[e] * N_HEADS + hd;
  float lg = leaky02(a_s[src[e] * N_HEADS + hd] + a_d[di]);
  atomicAdd(&s[di], expf(lg - m[di]));
}
__global__ void gat_alpha(const int* __restrict__ src, const int* __restrict__ dst,
                          const float* __restrict__ a_s, const float* __restrict__ a_d,
                          const float* __restrict__ m, const float* __restrict__ s,
                          float* __restrict__ alpha) {
  int idx = blockIdx.x * blockDim.x + threadIdx.x;
  if (idx >= N_EDGES * N_HEADS) return;
  int e = idx >> 2, hd = idx & 3;
  int di = dst[e] * N_HEADS + hd;
  float lg = leaky02(a_s[src[e] * N_HEADS + hd] + a_d[di]);
  alpha[idx] = expf(lg - m[di]) / (s[di] + 1e-16f);
}
__global__ void gat_y_init(const float* __restrict__ hg,
                           const float* __restrict__ a_s, const float* __restrict__ a_d,
                           const float* __restrict__ m, const float* __restrict__ s,
                           float* __restrict__ y) {
  int idx = blockIdx.x * blockDim.x + threadIdx.x;   // N * H
  if (idx >= N_NODES * H_DIM) return;
  int n = idx >> 6, c = idx & 63;
  float acc = 0.0f;
  for (int hd = 0; hd < N_HEADS; ++hd) {
    int ni = n * N_HEADS + hd;
    float lg = leaky02(a_s[ni] + a_d[ni]);
    float al = expf(lg - m[ni]) / (s[ni] + 1e-16f);
    acc += hg[(size_t)n * (N_HEADS * H_DIM) + hd * H_DIM + c] * al;
  }
  y[idx] = acc;   // sum over heads; /HEADS folded into BN scale
}
__global__ void gat_scatter(const int* __restrict__ src, const int* __restrict__ dst,
                            const float* __restrict__ hg,
                            const float* __restrict__ alpha,
                            float* __restrict__ y) {
  int idx = blockIdx.x * blockDim.x + threadIdx.x;   // E * (H/4)
  if (idx >= N_EDGES * (H_DIM / 4)) return;
  int e  = idx >> 4;
  int c4 = (idx & 15) << 2;
  int sN = src[e], dN = dst[e];
  float4 acc = make_float4(0.f, 0.f, 0.f, 0.f);
  for (int hd = 0; hd < N_HEADS; ++hd) {
    float al = alpha[e * N_HEADS + hd];
    float4 v = *(const float4*)(hg + (size_t)sN * (N_HEADS * H_DIM) + hd * H_DIM + c4);
    acc.x += v.x * al; acc.y += v.y * al; acc.z += v.z * al; acc.w += v.w * al;
  }
  float* yp = y + (size_t)dN * H_DIM + c4;
  atomicAdd(yp + 0, acc.x);
  atomicAdd(yp + 1, acc.y);
  atomicAdd(yp + 2, acc.z);
  atomicAdd(yp + 3, acc.w);
}

// ---------------------------------------------------------------------------
// Dual pooling + MLP head
// ---------------------------------------------------------------------------
__global__ void pool_init(float* psum, float* pmax, float* cnt) {
  int i = blockIdx.x * blockDim.x + threadIdx.x;
  if (i < G_GRAPH * H_DIM) { psum[i] = 0.0f; pmax[i] = -INFINITY; }
  if (i < G_GRAPH) cnt[i] = 0.0f;
}
__global__ void pool_accum(const float* __restrict__ h, const int* __restrict__ batch,
                           float* psum, float* pmax, float* cnt) {
  int idx = blockIdx.x * blockDim.x + threadIdx.x;
  if (idx >= N_NODES * H_DIM) return;
  int n = idx >> 6, c = idx & 63;
  int g = batch[n];
  float v = h[idx];
  atomicAdd(&psum[g * H_DIM + c], v);
  atomicMaxFloat(&pmax[g * H_DIM + c], v);
  if (c == 0) atomicAdd(&cnt[g], 1.0f);
}

__global__ void head_mlp(const float* __restrict__ psum, const float* __restrict__ pmax,
                         const float* __restrict__ cnt,
                         const float* __restrict__ c1W, const float* __restrict__ c1b,
                         const float* __restrict__ c2W, const float* __restrict__ c2b,
                         const float* __restrict__ c3W, const float* __restrict__ c3b,
                         float* __restrict__ out) {
  __shared__ float z[2 * H_DIM];
  __shared__ float z1[H_DIM];
  __shared__ float z2[H_DIM / 2];
  int g = blockIdx.x;
  int t = threadIdx.x;  // 128 threads
  if (t < H_DIM) {
    z[t] = psum[g * H_DIM + t] / fmaxf(cnt[g], 1.0f);
  } else {
    float mv = pmax[g * H_DIM + t - H_DIM];
    z[t] = isfinite(mv) ? mv : 0.0f;
  }
  __syncthreads();
  if (t < H_DIM) {
    float acc = c1b[t];
    for (int k = 0; k < 2 * H_DIM; ++k) acc += z[k] * c1W[k * H_DIM + t];
    z1[t] = fmaxf(acc, 0.0f);
  }
  __syncthreads();
  if (t < H_DIM / 2) {
    float acc = c2b[t];
    for (int k = 0; k < H_DIM; ++k) acc += z1[k] * c2W[k * (H_DIM / 2) + t];
    z2[t] = fmaxf(acc, 0.0f);
  }
  __syncthreads();
  if (t < N_CLS) {
    float acc = c3b[t];
    for (int k = 0; k < H_DIM / 2; ++k) acc += z2[k] * c3W[k * N_CLS + t];
    out[g * N_CLS + t] = acc;
  }
}

// ---------------------------------------------------------------------------
// Orchestration
// ---------------------------------------------------------------------------
static inline int cdiv(int a, int b) { return (a + b - 1) / b; }

extern "C" void kernel_launch(void* const* d_in, const int* in_sizes, int n_in,
                              void* d_out, int out_size, void* d_ws, size_t ws_size,
                              hipStream_t stream) {
  (void)in_sizes; (void)n_in; (void)out_size; (void)ws_size;
  const float* x        = (const float*)d_in[0];
  const int*   ei       = (const int*)d_in[1];
  const int*   batch    = (const int*)d_in[2];
  const float* Wi       = (const float*)d_in[3];
  const float* bi       = (const float*)d_in[4];
  const float* gcn_W    = (const float*)d_in[5];
  const float* gcn_b    = (const float*)d_in[6];
  const float* gcn_ga   = (const float*)d_in[7];
  const float* gcn_be   = (const float*)d_in[8];
  const float* Wg       = (const float*)d_in[9];
  const float* bg       = (const float*)d_in[10];
  const float* att_src  = (const float*)d_in[11];
  const float* att_dst  = (const float*)d_in[12];
  const float* gat_ga   = (const float*)d_in[13];
  const float* gat_be   = (const float*)d_in[14];
  const float* c1W = (const float*)d_in[15]; const float* c1b = (const float*)d_in[16];
  const float* c2W = (const float*)d_in[17]; const float* c2b = (const float*)d_in[18];
  const float* c3W = (const float*)d_in[19]; const float* c3b = (const float*)d_in[20];
  float* out = (float*)d_out;

  const int* src = ei;
  const int* dst = ei + N_EDGES;

  // workspace layout (floats)
  float* W = (float*)d_ws;
  size_t o = 0;
  float* h    = W + o; o += (size_t)N_NODES * H_DIM;
  float* y    = W + o; o += (size_t)N_NODES * H_DIM;
  float* dinv = W + o; o += N_NODES;
  float* a_s  = W + o; o += (size_t)N_NODES * N_HEADS;
  float* a_d  = W + o; o += (size_t)N_NODES * N_HEADS;
  float* mbuf = W + o; o += (size_t)N_NODES * N_HEADS;
  float* sbuf = W + o; o += (size_t)N_NODES * N_HEADS;
  float* stats = W + o; o += 2 * H_DIM;
  float* psum = W + o; o += G_GRAPH * H_DIM;
  float* pmax = W + o; o += G_GRAPH * H_DIM;
  float* cnt  = W + o; o += G_GRAPH;
  float* alpha = W + o; o += (size_t)N_EDGES * N_HEADS;
  float* big   = W + o; o += (size_t)N_NODES * N_HEADS * H_DIM; // hw (GCN) / hg (GAT)

  const int NH  = N_NODES * H_DIM;          // 3.2M
  const int NHd = N_NODES * N_HEADS;        // 200K
  const int EHd = N_EDGES * N_HEADS;        // 3.2M
  const int EC4 = N_EDGES * (H_DIM / 4);    // 12.8M

  // degrees (dst-only, reused by all GCN layers)
  deg_init    <<<cdiv(N_NODES, 256), 256, 0, stream>>>(dinv);
  deg_edges   <<<cdiv(N_EDGES, 256), 256, 0, stream>>>(dst, dinv);
  deg_finalize<<<cdiv(N_NODES, 256), 256, 0, stream>>>(dinv);

  // input projection: h = relu(x @ Wi + bi)
  {
    dim3 grid(cdiv(N_NODES / 16, 8), H_DIM / 16);
    gemm_wmma_kernel<<<grid, 256, 0, stream>>>(x, Wi, bi, h, N_NODES, F_IN, H_DIM, 1);
  }

  // GCN layers
  for (int l = 0; l < L_GCN; ++l) {
    dim3 grid(cdiv(N_NODES / 16, 8), H_DIM / 16);
    gemm_wmma_kernel<<<grid, 256, 0, stream>>>(h, gcn_W + (size_t)l * H_DIM * H_DIM,
                                               nullptr, big, N_NODES, H_DIM, H_DIM, 0);
    gcn_self_init   <<<cdiv(NH, 256), 256, 0, stream>>>(big, dinv, gcn_b + l * H_DIM, y);
    gcn_edge_scatter<<<cdiv(EC4, 256), 256, 0, stream>>>(src, dst, dinv, big, y);
    bn_stats_zero   <<<1, 128, 0, stream>>>(stats);
    bn_stats        <<<1024, 256, 0, stream>>>(y, 1.0f, nullptr, stats);
    bn_act_residual <<<cdiv(NH, 256), 256, 0, stream>>>(y, 1.0f, nullptr, stats,
                                                        gcn_ga + l * H_DIM,
                                                        gcn_be + l * H_DIM, h, 0);
  }

  // GAT layer
  {
    dim3 grid(cdiv(N_NODES / 16, 8), (N_HEADS * H_DIM) / 16);
    gemm_wmma_kernel<<<grid, 256, 0, stream>>>(h, Wg, nullptr, big,
                                               N_NODES, H_DIM, N_HEADS * H_DIM, 0);
  }
  gat_logits <<<cdiv(NHd, 256), 256, 0, stream>>>(big, att_src, att_dst, a_s, a_d);
  gat_m_init <<<cdiv(NHd, 256), 256, 0, stream>>>(a_s, a_d, mbuf);
  gat_m_edges<<<cdiv(EHd, 256), 256, 0, stream>>>(src, dst, a_s, a_d, mbuf);
  gat_s_init <<<cdiv(NHd, 256), 256, 0, stream>>>(a_s, a_d, mbuf, sbuf);
  gat_s_edges<<<cdiv(EHd, 256), 256, 0, stream>>>(src, dst, a_s, a_d, mbuf, sbuf);
  gat_alpha  <<<cdiv(EHd, 256), 256, 0, stream>>>(src, dst, a_s, a_d, mbuf, sbuf, alpha);
  gat_y_init <<<cdiv(NH, 256), 256, 0, stream>>>(big, a_s, a_d, mbuf, sbuf, y);
  gat_scatter<<<cdiv(EC4, 256), 256, 0, stream>>>(src, dst, big, alpha, y);
  bn_stats_zero  <<<1, 128, 0, stream>>>(stats);
  bn_stats       <<<1024, 256, 0, stream>>>(y, 1.0f / N_HEADS, bg, stats);
  bn_act_residual<<<cdiv(NH, 256), 256, 0, stream>>>(y, 1.0f / N_HEADS, bg, stats,
                                                     gat_ga, gat_be, h, 1);

  // pooling + head
  pool_init <<<cdiv(G_GRAPH * H_DIM, 256), 256, 0, stream>>>(psum, pmax, cnt);
  pool_accum<<<cdiv(NH, 256), 256, 0, stream>>>(h, batch, psum, pmax, cnt);
  head_mlp  <<<G_GRAPH, 128, 0, stream>>>(psum, pmax, cnt, c1W, c1b, c2W, c2b,
                                          c3W, c3b, out);
}
